// CasualSelfAttention_69526930588430
// MI455X (gfx1250) — matrix operand order repaired and checked by reading, not compiled, not run on hardware
//
#include <hip/hip_runtime.h>
#include <hip/hip_bf16.h>

typedef __bf16 bf16;
typedef __attribute__((ext_vector_type(16))) __bf16 v16bf;
typedef __attribute__((ext_vector_type(8)))  __bf16 v8bf;
typedef __attribute__((ext_vector_type(8)))  float  v8f;
typedef __attribute__((ext_vector_type(8)))  unsigned int v8u;

#define B_  2
#define T_  2048
#define C_  1024
#define H_  16
#define D_  64
#define N3_ 3072
#define KT_ (C_ / 32)        // 32 k-tiles of 32
#define SCALE_ 0.0125f       // 0.1 / sqrt(64)

// ---------------- WMMA + swizzled-fragment helpers --------------------------
// Fragment-major storage: each 16x32 (A: MxK) or 32x16 (B: KxN) fragment is a
// 512-element region; lane owns 16 contiguous bf16 (32B) at region + lane*16.

__device__ __forceinline__ v8f wmma_bf16(v16bf a, v16bf b, v8f c) {
  return __builtin_amdgcn_wmma_f32_16x16x32_bf16(false, a, false, b, (short)0, c,
                                                 false, false);
}

// A-frag element offset: lane = m | (((k>>4)&1)<<4) ; e = k & 15
__device__ __forceinline__ int swzA(int m, int k) {
  return ((m | (((k >> 4) & 1) << 4)) << 4) | (k & 15);
}
// B-frag element offset: lane = (n&15) | (((k>>4)&1)<<4) ; e = k & 15
__device__ __forceinline__ int swzB(int k, int n) {
  return (((n & 15) | (((k >> 4) & 1) << 4)) << 4) | (k & 15);
}

// 32B-per-lane fragment load (global or LDS pointer) -> 2x *_load_b128
__device__ __forceinline__ v16bf load_frag(const bf16* p) {
  return __builtin_bit_cast(v16bf, *(const v8u*)(p + ((threadIdx.x & 31) << 4)));
}

// ---------------- prep: swizzle fp32 operands into bf16 fragment layout -----

// x [4096][1024] fp32 -> A-frag tiles: region ((m>>4)*KT_ + (k>>5))*512
__global__ __launch_bounds__(256) void swz_x(const float* __restrict__ x,
                                             bf16* __restrict__ dst) {
  const int id = blockIdx.x * 256 + threadIdx.x;   // (B*T*C)/8 threads
  const int m  = id >> 7;                          // C_/8 = 128 segs per row
  const int k0 = (id & 127) << 3;
  const float* src = x + (size_t)m * C_ + k0;
  v8bf t;
#pragma unroll
  for (int i = 0; i < 8; ++i) t[i] = (bf16)src[i];
  bf16* out = dst + ((size_t)(m >> 4) * KT_ + (k0 >> 5)) * 512 + swzA(m & 15, k0 & 31);
  *(v8bf*)out = t;   // 16B aligned vector store
}

// w [K][N] fp32 -> B-frag tiles: region ((n>>4)*(K/32) + kt)*512
template <int N, int K>
__global__ __launch_bounds__(256) void swz_w(const float* __restrict__ w,
                                             bf16* __restrict__ dst) {
  const int id = blockIdx.x * 256 + threadIdx.x;   // N*(K/32) threads
  const int n  = id % N;
  const int kt = id / N;
  const float* src = w + (size_t)kt * 32 * N + n;  // column walk, lane-coalesced
  v16bf lo, hi;
#pragma unroll
  for (int j = 0; j < 16; ++j) lo[j] = (bf16)src[(size_t)j * N];
#pragma unroll
  for (int j = 0; j < 16; ++j) hi[j] = (bf16)src[(size_t)(16 + j) * N];
  bf16* base = dst + ((size_t)(n >> 4) * (K / 32) + kt) * 512;
  *(v16bf*)(base + ((n & 15) << 4))        = lo;   // lanes for k=0..15
  *(v16bf*)(base + (((n & 15) | 16) << 4)) = hi;   // lanes for k=16..31
}

// ---------------- Kernel 1: QKV GEMM (register-only, LDS-free) --------------
// WG 256 thr = 8 waves (2m x 4n), WG tile 64x256, wave tile 32x64.
// Epilogue scatters Q/K/V into the exact fragment layouts attention consumes.

__global__ __launch_bounds__(256) void qkv_gemm(
    const bf16* __restrict__ Xsw, const bf16* __restrict__ Wsw,
    const float* __restrict__ bias,
    bf16* __restrict__ Qsw, bf16* __restrict__ Ksw, bf16* __restrict__ Vsw) {
  const int tid = threadIdx.x;
  const int lane = tid & 31, wave = tid >> 5;
  const int wm = wave >> 2, wn = wave & 3;
  const int m0 = blockIdx.y * 64;
  const int n0 = blockIdx.x * 256;
  const int mt0 = (m0 >> 4) + wm * 2;
  const int nt0 = (n0 >> 4) + wn * 4;

  v8f acc[2][4] = {};
  for (int kt = 0; kt < KT_; ++kt) {
    const v16bf a0 = load_frag(Xsw + ((size_t)mt0 * KT_ + kt) * 512);
    const v16bf a1 = load_frag(Xsw + ((size_t)(mt0 + 1) * KT_ + kt) * 512);
#pragma unroll
    for (int j = 0; j < 4; ++j) {
      const v16bf b = load_frag(Wsw + ((size_t)(nt0 + j) * KT_ + kt) * 512);
      acc[0][j] = wmma_bf16(a0, b, acc[0][j]);
      acc[1][j] = wmma_bf16(a1, b, acc[1][j]);
    }
  }

  const int half = lane >> 4, nl = lane & 15;
  const int which = n0 >> 10;                     // block-uniform: 0=q 1=k 2=v
  const float sc = (which == 0) ? SCALE_ : 1.0f;
  bf16* dst = (which == 0) ? Qsw : (which == 1) ? Ksw : Vsw;
#pragma unroll
  for (int i = 0; i < 2; ++i)
#pragma unroll
    for (int j = 0; j < 4; ++j)
#pragma unroll
      for (int r = 0; r < 8; ++r) {
        const int gm = m0 + wm * 32 + i * 16 + r + 8 * half;
        const int gn = n0 + wn * 64 + j * 16 + nl;
        const float v = (acc[i][j][r] + bias[gn]) * sc;
        const int cc = gn & (C_ - 1);
        const int h = cc >> 6, d = cc & 63;
        const int b = gm >> 11, t = gm & (T_ - 1);
        const int bh = b * H_ + h;
        size_t idx;
        if (which == 0)        // Q: A-frag tiles per (bh, t-tile16) x 2 d-halves
          idx = ((size_t)bh * (T_ / 16) + (t >> 4)) * 1024 + (d >> 5) * 512 +
                swzA(t & 15, d & 31);
        else if (which == 1)   // K: B-frag (n=key16, k=d32) for S = Q K^T
          idx = (((size_t)bh * (T_ / 16) + (t >> 4)) * 2 + (d >> 5)) * 512 +
                swzB(d & 31, t & 15);
        else                   // V: B-frag (k=key32, n=d16) for P V
          idx = (((size_t)bh * (T_ / 32) + (t >> 5)) * 4 + (d >> 4)) * 512 +
                swzB(t & 31, d & 15);
        dst[idx] = (bf16)v;
      }
}

// ---------------- Kernel 2: causal flash attention (barrier-free) -----------
// 128 thr = 4 independent waves; wave owns 32 query rows (two 16-row tiles so
// K/V fragments are reused: 18 WMMA per 8 fragment loads per 32-key block).
// Row-sum of P comes from a WMMA against an all-ones fragment (no sum tree).
// Only LDS: per-wave 2KB P bounce (C-layout -> A-layout), wave-local fence.

__global__ __launch_bounds__(128) void attn_fwd(
    const bf16* __restrict__ Qsw, const bf16* __restrict__ Ksw,
    const bf16* __restrict__ Vsw, bf16* __restrict__ Ysw) {
  __shared__ bf16 Ps[4][1024];
  const int tid = threadIdx.x;
  const int lane = tid & 31, wave = tid >> 5;
  const int bh = blockIdx.x;
  const int b = bh >> 4, h = bh & (H_ - 1);
  const int qw = blockIdx.y * 128 + wave * 32;    // this wave's first query row
  const int half = lane >> 4, nl = lane & 15;

  const bf16* Kb = Ksw + (size_t)bh * (T_ / 16) * 1024;
  const bf16* Vb = Vsw + (size_t)bh * (T_ / 32) * 2048;
  const bf16* Qb = Qsw + ((size_t)bh * (T_ / 16) + (qw >> 4)) * 1024;
  v16bf qa[2][2];                                  // [row tile][d half]
#pragma unroll
  for (int mi = 0; mi < 2; ++mi) {
    qa[mi][0] = load_frag(Qb + (size_t)mi * 1024);
    qa[mi][1] = load_frag(Qb + (size_t)mi * 1024 + 512);
  }
  v16bf ones;
#pragma unroll
  for (int e = 0; e < 16; ++e) ones[e] = (bf16)1.0f;

  v8f o[2][4] = {};
  v8f lacc[2] = {};
  float mrow[2][8];
#pragma unroll
  for (int mi = 0; mi < 2; ++mi)
#pragma unroll
    for (int r = 0; r < 8; ++r) mrow[mi][r] = -1e30f;

  const int kbmax = qw >> 5;                       // inclusive causal bound
  for (int kb = 0; kb <= kbmax; ++kb) {
    const int kstart = kb << 5;
    // shared K fragments for both row tiles: [key tile][d half]
    const v16bf kf[2][2] = {
        {load_frag(Kb + ((size_t)(kb * 2) * 2 + 0) * 512),
         load_frag(Kb + ((size_t)(kb * 2) * 2 + 1) * 512)},
        {load_frag(Kb + ((size_t)(kb * 2 + 1) * 2 + 0) * 512),
         load_frag(Kb + ((size_t)(kb * 2 + 1) * 2 + 1) * 512)}};

#pragma unroll
    for (int mi = 0; mi < 2; ++mi) {
      // S = Q K^T : 16 rows x 32 keys (4 WMMA)
      v8f s0 = {}, s1 = {};
      s0 = wmma_bf16(qa[mi][0], kf[0][0], s0);
      s0 = wmma_bf16(qa[mi][1], kf[0][1], s0);
      s1 = wmma_bf16(qa[mi][0], kf[1][0], s1);
      s1 = wmma_bf16(qa[mi][1], kf[1][1], s1);

      // online softmax: causal mask, row max (tree), exp, rescale O and l
      float p0[8], p1[8], alpha[8];
#pragma unroll
      for (int r = 0; r < 8; ++r) {
        const int q = qw + mi * 16 + r + 8 * half;
        const float v0 = (kstart + nl      <= q) ? s0[r] : -1e30f;
        const float v1 = (kstart + 16 + nl <= q) ? s1[r] : -1e30f;
        float mv = fmaxf(v0, v1);
        mv = fmaxf(mv, __shfl_xor(mv, 1, 32));
        mv = fmaxf(mv, __shfl_xor(mv, 2, 32));
        mv = fmaxf(mv, __shfl_xor(mv, 4, 32));
        mv = fmaxf(mv, __shfl_xor(mv, 8, 32));
        const float mnew = fmaxf(mrow[mi][r], mv);
        alpha[r] = __expf(mrow[mi][r] - mnew);
        p0[r] = __expf(v0 - mnew);
        p1[r] = __expf(v1 - mnew);
        mrow[mi][r] = mnew;
      }
#pragma unroll
      for (int r = 0; r < 8; ++r) lacc[mi][r] *= alpha[r];
#pragma unroll
      for (int nt = 0; nt < 4; ++nt)
#pragma unroll
        for (int r = 0; r < 8; ++r) o[mi][nt][r] *= alpha[r];

      // P tile: C-layout regs -> A-frag LDS (per-wave region)
      bf16* pw = Ps[wave] + mi * 512;
#pragma unroll
      for (int r = 0; r < 8; ++r) {
        const int m = r + 8 * half;
        pw[swzA(m, nl)]      = (bf16)p0[r];
        pw[swzA(m, nl + 16)] = (bf16)p1[r];
      }
    }
    asm volatile("s_wait_dscnt 0" ::: "memory");

    // shared V fragments, reused by both row tiles; l via P x ones
    const v16bf vb0 = load_frag(Vb + ((size_t)kb * 4 + 0) * 512);
    const v16bf vb1 = load_frag(Vb + ((size_t)kb * 4 + 1) * 512);
    const v16bf vb2 = load_frag(Vb + ((size_t)kb * 4 + 2) * 512);
    const v16bf vb3 = load_frag(Vb + ((size_t)kb * 4 + 3) * 512);
#pragma unroll
    for (int mi = 0; mi < 2; ++mi) {
      const v16bf pa = load_frag(Ps[wave] + mi * 512);
      o[mi][0] = wmma_bf16(pa, vb0, o[mi][0]);
      o[mi][1] = wmma_bf16(pa, vb1, o[mi][1]);
      o[mi][2] = wmma_bf16(pa, vb2, o[mi][2]);
      o[mi][3] = wmma_bf16(pa, vb3, o[mi][3]);
      lacc[mi] = wmma_bf16(pa, ones, lacc[mi]);    // row sums, all lanes
    }
  }

  // normalize, store Y pre-swizzled as proj's A operand
#pragma unroll
  for (int mi = 0; mi < 2; ++mi) {
    float inv[8];
#pragma unroll
    for (int r = 0; r < 8; ++r) inv[r] = 1.0f / lacc[mi][r];
#pragma unroll
    for (int nt = 0; nt < 4; ++nt)
#pragma unroll
      for (int r = 0; r < 8; ++r) {
        const int t = qw + mi * 16 + r + 8 * half;
        const int c = h * D_ + nt * 16 + nl;
        const int gm = b * T_ + t;
        Ysw[((size_t)(gm >> 4) * KT_ + (c >> 5)) * 512 + swzA(gm & 15, c & 31)] =
            (bf16)(o[mi][nt][r] * inv[r]);
      }
  }
}

// ---------------- Kernel 3: out = Y @ w_proj + b_proj (fp32, LDS-free) ------

__global__ __launch_bounds__(256) void proj_gemm(
    const bf16* __restrict__ Ysw, const bf16* __restrict__ Wsw,
    const float* __restrict__ bias, float* __restrict__ out) {
  const int tid = threadIdx.x;
  const int lane = tid & 31, wave = tid >> 5;
  const int wm = wave >> 2, wn = wave & 3;
  const int m0 = blockIdx.y * 64;
  const int n0 = blockIdx.x * 256;
  const int mt0 = (m0 >> 4) + wm * 2;
  const int nt0 = (n0 >> 4) + wn * 4;

  v8f acc[2][4] = {};
  for (int kt = 0; kt < KT_; ++kt) {
    const v16bf a0 = load_frag(Ysw + ((size_t)mt0 * KT_ + kt) * 512);
    const v16bf a1 = load_frag(Ysw + ((size_t)(mt0 + 1) * KT_ + kt) * 512);
#pragma unroll
    for (int j = 0; j < 4; ++j) {
      const v16bf b = load_frag(Wsw + ((size_t)(nt0 + j) * KT_ + kt) * 512);
      acc[0][j] = wmma_bf16(a0, b, acc[0][j]);
      acc[1][j] = wmma_bf16(a1, b, acc[1][j]);
    }
  }

  const int half = lane >> 4, nl = lane & 15;
#pragma unroll
  for (int i = 0; i < 2; ++i)
#pragma unroll
    for (int j = 0; j < 4; ++j)
#pragma unroll
      for (int r = 0; r < 8; ++r) {
        const int gm = m0 + wm * 32 + i * 16 + r + 8 * half;
        const int gn = n0 + wn * 64 + j * 16 + nl;
        out[(size_t)gm * C_ + gn] = acc[i][j][r] + bias[gn];
      }
}

// ---------------- launch ----------------------------------------------------

extern "C" void kernel_launch(void* const* d_in, const int* in_sizes, int n_in,
                              void* d_out, int out_size, void* d_ws,
                              size_t ws_size, hipStream_t stream) {
  const float* x      = (const float*)d_in[0];
  const float* w_attn = (const float*)d_in[1];
  const float* b_attn = (const float*)d_in[2];
  const float* w_proj = (const float*)d_in[3];
  const float* b_proj = (const float*)d_in[4];
  float* out = (float*)d_out;

  // workspace (bf16 elems): Xsw 4M, Wa 3M, Wp 1M, Q/K/V 4M each, Y 4M = 48MB
  const size_t nX  = (size_t)B_ * T_ * C_;
  const size_t nWa = (size_t)C_ * N3_;
  const size_t nWp = (size_t)C_ * C_;
  const size_t nQ  = (size_t)B_ * H_ * T_ * D_;
  bf16* Xsw  = (bf16*)d_ws;
  bf16* Wasw = Xsw + nX;
  bf16* Wpsw = Wasw + nWa;
  bf16* Qsw  = Wpsw + nWp;
  bf16* Ksw  = Qsw + nQ;
  bf16* Vsw  = Ksw + nQ;
  bf16* Ysw  = Vsw + nQ;

  swz_x<<<(B_ * T_ * C_ / 8) / 256, 256, 0, stream>>>(x, Xsw);
  swz_w<N3_, C_><<<(N3_ * KT_) / 256, 256, 0, stream>>>(w_attn, Wasw);
  swz_w<C_, C_><<<(C_ * KT_) / 256, 256, 0, stream>>>(w_proj, Wpsw);

  qkv_gemm<<<dim3(N3_ / 256, (B_ * T_) / 64), 256, 0, stream>>>(
      Xsw, Wasw, b_attn, Qsw, Ksw, Vsw);
  attn_fwd<<<dim3(B_ * H_, T_ / 128), 128, 0, stream>>>(Qsw, Ksw, Vsw, Ysw);
  proj_gemm<<<dim3(C_ / 256, (B_ * T_) / 64), 256, 0, stream>>>(
      Ysw, Wpsw, b_proj, out);
}
